// MLP_56779467653689
// MI455X (gfx1250) — matrix operand order repaired and checked by reading, not compiled, not run on hardware
//
#include <hip/hip_runtime.h>

// ---------------------------------------------------------------------------
// Binary-weight MLP on MI455X (gfx1250, wave32, WMMA).
//   h0 = relu(s0 * x @ W0) * 2 ;  h1 = relu(s1 * h0 @ W1) * 2 ;  out = s2 * h1 @ W2
// Weights are +/-1 -> exact in bf16. Activations bf16, accumulate f32 via
// V_WMMA_F32_16X16X32_BF16. Weights pre-transposed to [N][K] so B fragments
// are contiguous LDS reads. Tile staging uses CDNA5 async global->LDS loads
// (ASYNCcnt) when the toolchain exposes them; otherwise reg-staged copies.
// ---------------------------------------------------------------------------

typedef __attribute__((ext_vector_type(16))) __bf16       v16bf;
typedef __attribute__((ext_vector_type(8)))  float        v8f;
typedef __attribute__((ext_vector_type(4)))  unsigned int u32x4;
typedef __attribute__((ext_vector_type(4)))  int          v4i;

#define TOKENS 8192
#define D0 1024
#define D1 4096
#define D2 4096
#define D3 1024

// GEMM tiling
#define BM 128
#define BN 128
#define BK 32
#define KP 40                 // padded LDS row stride in halves (80B, 16B-mult)
#define WAVES_N 4
#define WM 64                 // wave tile M (2 waves in M)
#define WN 32                 // wave tile N (4 waves in N)
#define MI 4                  // WM/16
#define NI 2                  // WN/16

#if __has_builtin(__builtin_amdgcn_global_load_async_to_lds_b128)
#define HAVE_ASYNC_LDS 1
// builtin expects: (AS1 v4i* src_global, AS3 v4i* dst_lds, imm offset, imm cpol)
typedef __attribute__((address_space(1))) v4i as1_v4i;
typedef __attribute__((address_space(3))) v4i as3_v4i;
#define ASYNC_CP_B128(gsrc, ldst) \
  __builtin_amdgcn_global_load_async_to_lds_b128((as1_v4i*)(gsrc), \
                                                 (as3_v4i*)(ldst), 0, 0)
#else
#define HAVE_ASYNC_LDS 0
#endif

#if __has_builtin(__builtin_amdgcn_s_wait_asynccnt)
#define WAIT_ASYNC0() __builtin_amdgcn_s_wait_asynccnt(0)
#elif HAVE_ASYNC_LDS
#define WAIT_ASYNC0() asm volatile("s_wait_asynccnt 0" ::: "memory")
#else
#define WAIT_ASYNC0()
#endif

union FragCast { u32x4 u[2]; v16bf v; };

__device__ __forceinline__ unsigned short f32_to_bf16(float f) {
  unsigned int u = __float_as_uint(f);
  return (unsigned short)((u + 0x7FFFu + ((u >> 16) & 1u)) >> 16);
}

// f32 -> bf16 (RNE), elementwise
__global__ void xconvert_kernel(const float* __restrict__ x,
                                unsigned short* __restrict__ out, int n) {
  int i = blockIdx.x * blockDim.x + threadIdx.x;
  int stride = gridDim.x * blockDim.x;
  for (; i < n; i += stride) out[i] = f32_to_bf16(x[i]);
}

// bool mask [K][N] (row-major, int) -> bf16 +/-1 transposed to [N][K]
__global__ void wconvert_kernel(const int* __restrict__ k,
                                unsigned short* __restrict__ wt,
                                int K, int N) {
  __shared__ unsigned short tile[32][33];
  const int bn = blockIdx.x * 32;
  const int bk = blockIdx.y * 32;
  const int tx = threadIdx.x, ty = threadIdx.y;
  for (int i = ty; i < 32; i += 8)
    tile[i][tx] = k[(size_t)(bk + i) * N + (bn + tx)] ? 0x3F80u : 0xBF80u;
  __syncthreads();
  for (int i = ty; i < 32; i += 8)
    wt[(size_t)(bn + i) * K + (bk + tx)] = tile[tx][i];
}

// C[M,N] = epilogue( A[M,K](bf16) @ Bt[N,K]^T(bf16) )
//   RELU=true : bf16 out = max(acc,0) * (2*scale)
//   RELU=false: f32  out = acc * scale
template <bool RELU>
__global__ __launch_bounds__(256) void gemm_bin_kernel(
    const unsigned short* __restrict__ A,   // [M][K] bf16 bits
    const unsigned short* __restrict__ Bt,  // [N][K] bf16 bits (pre-transposed)
    const float* __restrict__ scale,
    void* __restrict__ Cout, int M, int N, int K) {
  __shared__ __align__(16) unsigned short As[2][BM * KP];
  __shared__ __align__(16) unsigned short Bs[2][BN * KP];

  const int tid  = threadIdx.x;
  const int lane = tid & 31;
  const int wave = tid >> 5;
  const int wm   = (wave / WAVES_N) * WM;
  const int wn   = (wave % WAVES_N) * WN;
  const int m0   = blockIdx.y * BM;
  const int n0   = blockIdx.x * BN;

  // Cooperative tile fill: 128 rows x 4 chunks (8 halves each) = 512 chunks,
  // 2 chunks per thread.  chunk c: row = c>>2, koff = (c&3)*8
  const int r0 = (tid)       >> 2, o0 = ((tid)       & 3) * 8;
  const int r1 = (tid + 256) >> 2, o1 = ((tid + 256) & 3) * 8;

  v8f acc[MI][NI] = {};
  const int nk = K / BK;

#if HAVE_ASYNC_LDS
  // ---- async global->LDS staging (no VGPR round-trip) ----
  ASYNC_CP_B128(A  + (size_t)(m0 + r0) * K + o0, &As[0][r0 * KP + o0]);
  ASYNC_CP_B128(A  + (size_t)(m0 + r1) * K + o1, &As[0][r1 * KP + o1]);
  ASYNC_CP_B128(Bt + (size_t)(n0 + r0) * K + o0, &Bs[0][r0 * KP + o0]);
  ASYNC_CP_B128(Bt + (size_t)(n0 + r1) * K + o1, &Bs[0][r1 * KP + o1]);
  WAIT_ASYNC0();
  __syncthreads();
#else
  // ---- fallback: reg-staged copy ----
  u32x4 ga0 = *(const u32x4*)(A  + (size_t)(m0 + r0) * K + o0);
  u32x4 ga1 = *(const u32x4*)(A  + (size_t)(m0 + r1) * K + o1);
  u32x4 gb0 = *(const u32x4*)(Bt + (size_t)(n0 + r0) * K + o0);
  u32x4 gb1 = *(const u32x4*)(Bt + (size_t)(n0 + r1) * K + o1);
  *(u32x4*)&As[0][r0 * KP + o0] = ga0;
  *(u32x4*)&As[0][r1 * KP + o1] = ga1;
  *(u32x4*)&Bs[0][r0 * KP + o0] = gb0;
  *(u32x4*)&Bs[0][r1 * KP + o1] = gb1;
  __syncthreads();
#endif

  const int am  = lane & 15;            // M (A) / N (B) within fragment
  const int alo = (lane < 16) ? 0 : 8;  // A: K sub-block select per lane half

  for (int kt = 0; kt < nk; ++kt) {
    const int cur = kt & 1;

    if (kt + 1 < nk) {  // start fill of the other buffer
      const int kb  = (kt + 1) * BK;
      const int nxt = cur ^ 1;
#if HAVE_ASYNC_LDS
      ASYNC_CP_B128(A  + (size_t)(m0 + r0) * K + kb + o0, &As[nxt][r0 * KP + o0]);
      ASYNC_CP_B128(A  + (size_t)(m0 + r1) * K + kb + o1, &As[nxt][r1 * KP + o1]);
      ASYNC_CP_B128(Bt + (size_t)(n0 + r0) * K + kb + o0, &Bs[nxt][r0 * KP + o0]);
      ASYNC_CP_B128(Bt + (size_t)(n0 + r1) * K + kb + o1, &Bs[nxt][r1 * KP + o1]);
#else
      ga0 = *(const u32x4*)(A  + (size_t)(m0 + r0) * K + kb + o0);
      ga1 = *(const u32x4*)(A  + (size_t)(m0 + r1) * K + kb + o1);
      gb0 = *(const u32x4*)(Bt + (size_t)(n0 + r0) * K + kb + o0);
      gb1 = *(const u32x4*)(Bt + (size_t)(n0 + r1) * K + kb + o1);
      (void)nxt;
#endif
    }
    if (kt + 2 < nk) {  // prefetch K+2 tile into L2 (global_prefetch_b8)
      const int kb2 = (kt + 2) * BK;
      __builtin_prefetch(A  + (size_t)(m0 + r0) * K + kb2 + o0, 0, 1);
      __builtin_prefetch(Bt + (size_t)(n0 + r0) * K + kb2 + o0, 0, 1);
    }

    // A fragments: lane = M row; lanes 0-15 hold K {0..7,16..23},
    //              lanes 16-31 hold K {8..15,24..31}   (ISA 7.12.2)
    v16bf afrag[MI], bfrag[NI];
#pragma unroll
    for (int mi = 0; mi < MI; ++mi) {
      const unsigned short* p = &As[cur][(wm + mi * 16 + am) * KP + alo];
      FragCast f;
      f.u[0] = *(const u32x4*)(p);
      f.u[1] = *(const u32x4*)(p + 16);
      afrag[mi] = f.v;
    }
    // B fragments: lane = N col; lanes 0-15 hold K {0..7,16..23}, lanes 16-31 K {8..15,24..31}
#pragma unroll
    for (int ni = 0; ni < NI; ++ni) {
      const unsigned short* p = &Bs[cur][(wn + ni * 16 + am) * KP + alo];
      FragCast f;
      f.u[0] = *(const u32x4*)(p);
      f.u[1] = *(const u32x4*)(p + 16);
      bfrag[ni] = f.v;
    }

#pragma unroll
    for (int mi = 0; mi < MI; ++mi)
#pragma unroll
      for (int ni = 0; ni < NI; ++ni)
        acc[mi][ni] = __builtin_amdgcn_wmma_f32_16x16x32_bf16(
            false, afrag[mi], false, bfrag[ni], (short)0, acc[mi][ni],
            false, false);

    if (kt + 1 < nk) {
#if HAVE_ASYNC_LDS
      WAIT_ASYNC0();  // next buffer landed in LDS
#else
      const int nxt = cur ^ 1;
      *(u32x4*)&As[nxt][r0 * KP + o0] = ga0;
      *(u32x4*)&As[nxt][r1 * KP + o1] = ga1;
      *(u32x4*)&Bs[nxt][r0 * KP + o0] = gb0;
      *(u32x4*)&Bs[nxt][r1 * KP + o1] = gb1;
#endif
    }
    __syncthreads();
  }

  // Epilogue. C/D layout: lanes 0-15 -> M=r, N=lane; lanes 16-31 -> M=8+r.
  const float sc  = scale[0] * (RELU ? 2.0f : 1.0f);
  const int mbase = m0 + wm + ((lane < 16) ? 0 : 8);
  const int nbase = n0 + wn + am;
#pragma unroll
  for (int mi = 0; mi < MI; ++mi) {
#pragma unroll
    for (int ni = 0; ni < NI; ++ni) {
      const int n = nbase + ni * 16;
#pragma unroll
      for (int r = 0; r < 8; ++r) {
        const int m = mbase + mi * 16 + r;
        const float v = acc[mi][ni][r];
        if (RELU) {
          ((unsigned short*)Cout)[(size_t)m * N + n] =
              f32_to_bf16(fmaxf(v, 0.0f) * sc);
        } else {
          ((float*)Cout)[(size_t)m * N + n] = v * sc;
        }
      }
    }
  }
}

extern "C" void kernel_launch(void* const* d_in, const int* in_sizes, int n_in,
                              void* d_out, int out_size, void* d_ws,
                              size_t ws_size, hipStream_t stream) {
  const float* x  = (const float*)d_in[0];
  const int*   k0 = (const int*)d_in[1];
  const int*   k1 = (const int*)d_in[2];
  const int*   k2 = (const int*)d_in[3];
  const float* s0 = (const float*)d_in[4];
  const float* s1 = (const float*)d_in[5];
  const float* s2 = (const float*)d_in[6];

  char* ws = (char*)d_ws;
  size_t off = 0;
  unsigned short* W0t = (unsigned short*)(ws + off); off += (size_t)D1 * D0 * 2;  //  8 MB [4096][1024]
  unsigned short* W1t = (unsigned short*)(ws + off); off += (size_t)D2 * D1 * 2;  // 32 MB [4096][4096]
  unsigned short* W2t = (unsigned short*)(ws + off); off += (size_t)D3 * D2 * 2;  //  8 MB [1024][4096]
  unsigned short* Xb  = (unsigned short*)(ws + off); off += (size_t)TOKENS * D0 * 2;
  unsigned short* H0  = (unsigned short*)(ws + off); off += (size_t)TOKENS * D1 * 2;
  unsigned short* H1  = (unsigned short*)(ws + off); off += (size_t)TOKENS * D2 * 2;

  // weight conversion (+transpose to [N][K]) and activation conversion
  wconvert_kernel<<<dim3(D1 / 32, D0 / 32), dim3(32, 8), 0, stream>>>(k0, W0t, D0, D1);
  wconvert_kernel<<<dim3(D2 / 32, D1 / 32), dim3(32, 8), 0, stream>>>(k1, W1t, D1, D2);
  wconvert_kernel<<<dim3(D3 / 32, D2 / 32), dim3(32, 8), 0, stream>>>(k2, W2t, D2, D3);
  xconvert_kernel<<<2048, 256, 0, stream>>>(x, Xb, TOKENS * D0);

  // 3 fused GEMMs
  gemm_bin_kernel<true><<<dim3(D1 / BN, TOKENS / BM), 256, 0, stream>>>(
      Xb, W0t, s0, H0, TOKENS, D1, D0);
  gemm_bin_kernel<true><<<dim3(D2 / BN, TOKENS / BM), 256, 0, stream>>>(
      H0, W1t, s1, H1, TOKENS, D2, D1);
  gemm_bin_kernel<false><<<dim3(D3 / BN, TOKENS / BM), 256, 0, stream>>>(
      H1, W2t, s2, d_out, TOKENS, D3, D2);
}
